// Block_30700426231923
// MI455X (gfx1250) — matrix operand order repaired and checked by reading, not compiled, not run on hardware
//
#include <hip/hip_runtime.h>

// ---------------- problem constants (from setup_inputs) ----------------
constexpr int kB   = 2;
constexpr int kT   = 512;
constexpr int kC   = 768;
constexpr int kNH  = 12;
constexpr int kHS  = 64;          // kC / kNH
constexpr int kM   = kB * kT;     // 1024 rows
constexpr int k3C  = 3 * kC;      // 2304
constexpr float kScale = 0.125f;  // 1/sqrt(64)
constexpr float kClamp = 1000000000.0f;

typedef __bf16 bf16_t;
typedef __attribute__((ext_vector_type(16))) __bf16 v16bf;
typedef __attribute__((ext_vector_type(8)))  float  v8f;

static __device__ __forceinline__ v8f vzero8() {
  v8f z = {0.f, 0.f, 0.f, 0.f, 0.f, 0.f, 0.f, 0.f};
  return z;
}

// A-operand style load: two 16B chunks (8 bf16 each)
static __device__ __forceinline__ v16bf ld_pair(const bf16_t* p0, const bf16_t* p1) {
  v16bf r;
  ((uint4*)&r)[0] = *(const uint4*)p0;
  ((uint4*)&r)[1] = *(const uint4*)p1;
  return r;
}
// B-operand style load: two 16B chunks at p and p+16 (8 bf16 each)
static __device__ __forceinline__ v16bf ld_contig(const bf16_t* p) {
  v16bf r;
  ((uint4*)&r)[0] = *(const uint4*)p;
  ((uint4*)&r)[1] = *(const uint4*)(p + 16);
  return r;
}

static __device__ __forceinline__ v8f wmma_bf16(v16bf a, v16bf b, v8f c) {
  // D = A(16x32 bf16) x B(32x16 bf16) + C(16x16 f32)
  return __builtin_amdgcn_wmma_f32_16x16x32_bf16(
      /*neg_a=*/false, a, /*neg_b=*/false, b,
      /*c_mod=*/(short)0, c, /*reuse_a=*/false, /*reuse_b=*/false);
}

// ---------------- tiny conversion kernels ----------------
__global__ __launch_bounds__(256) void cvt_kernel(const float* __restrict__ s,
                                                  bf16_t* __restrict__ d, int n) {
  int i = blockIdx.x * 256 + threadIdx.x;
  if (i < n) d[i] = (bf16_t)s[i];
}

__global__ __launch_bounds__(256) void cvt_abs_kernel(const float* __restrict__ s,
                                                      bf16_t* __restrict__ d,
                                                      bf16_t* __restrict__ da, int n) {
  int i = blockIdx.x * 256 + threadIdx.x;
  if (i < n) {
    float v = s[i];
    d[i]  = (bf16_t)v;
    da[i] = (bf16_t)fabsf(v);
  }
}

// ---------------- QKV GEMM: center = x@Wt + b, radius = e@|W|t ----------------
// wave tile: 16(M) x 64(N); block = 4 waves -> 64x64; grid = (M/64)*(N/64).
// K-loop unrolled by 64 with ping-pong operand buffers: no register copies,
// loads for one buffer overlap the WMMAs consuming the other.
__global__ __launch_bounds__(128) void gemm_qkv_kernel(
    const bf16_t* __restrict__ xb, const bf16_t* __restrict__ eb,
    const bf16_t* __restrict__ wb, const bf16_t* __restrict__ wab,
    const float* __restrict__ bias,
    float* __restrict__ Cc, float* __restrict__ Rr) {
  const int lane = threadIdx.x & 31;
  const int wv   = threadIdx.x >> 5;
  const int mb   = blockIdx.x % (kM / 64);
  const int nb   = blockIdx.x / (kM / 64);
  const int m0   = mb * 64 + wv * 16;
  const int n0   = nb * 64;
  const int n15  = lane & 15;
  const int d0a  = (lane & 16) ? 8 : 0;
  const int d0b  = (lane & 16) ? 8 : 0;
  const int mrow = m0 + n15;

  const bf16_t* xrow = xb + (size_t)mrow * kC + d0a;
  const bf16_t* erow = eb + (size_t)mrow * kC + d0a;
  const bf16_t* wrow[4];
  const bf16_t* warow[4];
#pragma unroll
  for (int nblk = 0; nblk < 4; nblk++) {
    int ncol = n0 + nblk * 16 + n15;
    wrow[nblk]  = wb  + (size_t)ncol * kC + d0b;
    warow[nblk] = wab + (size_t)ncol * kC + d0b;
  }

  v8f accC[4], accR[4];
#pragma unroll
  for (int i = 0; i < 4; i++) { accC[i] = vzero8(); accR[i] = vzero8(); }

  // prologue: K-block 0 -> buffer 0
  v16bf ax0 = ld_pair(xrow, xrow + 16);
  v16bf ae0 = ld_pair(erow, erow + 16);
  v16bf bw0[4], ba0[4], ax1, ae1, bw1[4], ba1[4];
#pragma unroll
  for (int nblk = 0; nblk < 4; nblk++) {
    bw0[nblk] = ld_contig(wrow[nblk]);
    ba0[nblk] = ld_contig(warow[nblk]);
  }

  for (int kb = 0; kb < kC; kb += 64) {
    const int k1 = kb + 32, k2 = kb + 64;
    // load k1 -> buffer 1 (overlaps buffer-0 WMMAs below)
    ax1 = ld_pair(xrow + k1, xrow + k1 + 16);
    ae1 = ld_pair(erow + k1, erow + k1 + 16);
#pragma unroll
    for (int nblk = 0; nblk < 4; nblk++) {
      bw1[nblk] = ld_contig(wrow[nblk] + k1);
      ba1[nblk] = ld_contig(warow[nblk] + k1);
    }
    __builtin_prefetch(wrow[0] + k2, 0, 1);   // gfx1250 global_prefetch path
    __builtin_prefetch(warow[0] + k2, 0, 1);
    // consume buffer 0
#pragma unroll
    for (int nblk = 0; nblk < 4; nblk++) {
      accC[nblk] = wmma_bf16(ax0, bw0[nblk], accC[nblk]);
      accR[nblk] = wmma_bf16(ae0, ba0[nblk], accR[nblk]);
    }
    // load k2 -> buffer 0 (overlaps buffer-1 WMMAs below)
    if (k2 < kC) {
      ax0 = ld_pair(xrow + k2, xrow + k2 + 16);
      ae0 = ld_pair(erow + k2, erow + k2 + 16);
#pragma unroll
      for (int nblk = 0; nblk < 4; nblk++) {
        bw0[nblk] = ld_contig(wrow[nblk] + k2);
        ba0[nblk] = ld_contig(warow[nblk] + k2);
      }
    }
    // consume buffer 1
#pragma unroll
    for (int nblk = 0; nblk < 4; nblk++) {
      accC[nblk] = wmma_bf16(ax1, bw1[nblk], accC[nblk]);
      accR[nblk] = wmma_bf16(ae1, ba1[nblk], accR[nblk]);
    }
  }

  const int rbase = (lane >> 4) * 8;
#pragma unroll
  for (int nblk = 0; nblk < 4; nblk++) {
    int col = n0 + nblk * 16 + n15;
    float bv = bias[col];
#pragma unroll
    for (int r = 0; r < 8; r++) {
      int row = m0 + rbase + r;
      Cc[(size_t)row * k3C + col] = accC[nblk][r] + bv;
      Rr[(size_t)row * k3C + col] = accR[nblk][r];
    }
  }
}

// ---------------- repack: head-major layouts + v transposes ----------------
__global__ __launch_bounds__(256) void repack_kernel(
    const float* __restrict__ Cc, const float* __restrict__ Rr,
    bf16_t* __restrict__ qb, float* __restrict__ ql, float* __restrict__ qu,
    bf16_t* __restrict__ kbq, float* __restrict__ kl, float* __restrict__ ku,
    bf16_t* __restrict__ vT, bf16_t* __restrict__ vlpT, bf16_t* __restrict__ vlnT,
    bf16_t* __restrict__ vupT, bf16_t* __restrict__ vunT) {
  int i = blockIdx.x * 256 + threadIdx.x;
  if (i >= kM * k3C) return;
  int m = i / k3C, n = i % k3C;
  int region = n / kC;
  int h = (n % kC) / kHS;
  int d = n % kHS;
  int b = m / kT, t = m % kT;
  int bh = b * kNH + h;
  size_t base  = ((size_t)bh * kT + t) * kHS + d;   // [B,H,T,HS]
  size_t baseT = ((size_t)bh * kHS + d) * kT + t;   // [B,H,HS,T]
  float c = Cc[i], r = Rr[i];
  float lo = c - r, hi = c + r;
  if (region == 0) {
    qb[base] = (bf16_t)c; ql[base] = lo; qu[base] = hi;
  } else if (region == 1) {
    kbq[base] = (bf16_t)c; kl[base] = lo; ku[base] = hi;
  } else {
    vT[baseT]   = (bf16_t)c;
    vlpT[baseT] = (bf16_t)fmaxf(lo, 0.f);
    vlnT[baseT] = (bf16_t)fminf(lo, 0.f);
    vupT[baseT] = (bf16_t)fmaxf(hi, 0.f);
    vunT[baseT] = (bf16_t)fminf(hi, 0.f);
  }
}

// ---------------- fused causal attention: std + interval, one pass ----------------
// ONE wave per workgroup (block=32): causal trip counts differ per q-tile, so
// single-wave workgroups avoid intra-block imbalance. grid = (B*H)*(T/16).
__global__ __launch_bounds__(32) void attn_kernel(
    const bf16_t* __restrict__ qb, const bf16_t* __restrict__ kbq,
    const float* __restrict__ ql, const float* __restrict__ qu,
    const float* __restrict__ kl, const float* __restrict__ ku,
    const bf16_t* __restrict__ vT, const bf16_t* __restrict__ vlpT,
    const bf16_t* __restrict__ vlnT, const bf16_t* __restrict__ vupT,
    const bf16_t* __restrict__ vunT,
    bf16_t* __restrict__ yb, bf16_t* __restrict__ ycb, bf16_t* __restrict__ yrb) {
  __shared__ bf16_t ldsS[16][32];
  __shared__ bf16_t ldsL[16][32];
  __shared__ bf16_t ldsU[16][32];

  const int lane = threadIdx.x & 31;
  const int bh   = blockIdx.x / (kT / 16);
  const int q0   = (blockIdx.x % (kT / 16)) * 16;
  const int hb   = lane >> 4;
  const int n15  = lane & 15;
  const int d0a  = hb ? 8 : 0;
  const int d0b  = hb ? 8 : 0;

  // q A-operands for K = 0..31 and 32..63 (constant across k-loop)
  const bf16_t* qrow = qb + ((size_t)bh * kT + q0 + n15) * kHS;
  v16bf aq0 = ld_pair(qrow + d0a, qrow + d0a + 16);
  v16bf aq1 = ld_pair(qrow + 32 + d0a, qrow + 32 + d0a + 16);

  // interval q-row bounds cached in registers (row n15, all 64 dims)
  const int qg_lane = q0 + n15;
  const float4* qlr4 = (const float4*)(ql + ((size_t)bh * kT + qg_lane) * kHS);
  const float4* qur4 = (const float4*)(qu + ((size_t)bh * kT + qg_lane) * kHS);
  float4 qlv[16], quv[16];
#pragma unroll
  for (int d4 = 0; d4 < 16; d4++) { qlv[d4] = qlr4[d4]; quv[d4] = qur4[d4]; }

  v8f Ny[4], N1[4], N2[4], N3[4], N4[4];
#pragma unroll
  for (int i = 0; i < 4; i++) {
    Ny[i] = vzero8(); N1[i] = vzero8(); N2[i] = vzero8();
    N3[i] = vzero8(); N4[i] = vzero8();
  }
  float dS = 0.f, dL = 0.f, dU = 0.f;

  const int kmax = q0 + 16;  // causal: last k needed is q0+15
  for (int kb = 0; kb < kmax; kb += 32) {
    // ---- standard scores: two 16x16 subtiles via WMMA (K = HS = 64) ----
#pragma unroll
    for (int sub = 0; sub < 2; sub++) {
      int kcb = kb + sub * 16;
      const bf16_t* krow = kbq + ((size_t)bh * kT + kcb + n15) * kHS;
      v16bf bk0 = ld_contig(krow + d0b);
      v16bf bk1 = ld_contig(krow + 32 + d0b);
      v8f s = vzero8();
      s = wmma_bf16(aq0, bk0, s);
      s = wmma_bf16(aq1, bk1, s);
      int kg = kcb + n15;
#pragma unroll
      for (int r = 0; r < 8; r++) {
        int row = r + hb * 8;
        int qg  = q0 + row;
        float e = (kg <= qg) ? __expf(s[r] * kScale) : 0.f;
        ldsS[row][sub * 16 + n15] = (bf16_t)e;
      }
    }

    // ---- issue v-tile loads EARLY: the interval VALU section below (~900
    // ops) hides their latency; the asm memory barrier later would otherwise
    // pin them to just before the WMMAs that consume them. ----
    v16bf bv[4], blp[4], bln[4], bup[4], bun[4];
#pragma unroll
    for (int hsb = 0; hsb < 4; hsb++) {
      size_t tb = ((size_t)bh * kHS + hsb * 16 + n15) * kT + kb + d0b;
      bv[hsb]  = ld_contig(vT + tb);
      blp[hsb] = ld_contig(vlpT + tb);
      bln[hsb] = ld_contig(vlnT + tb);
      bup[hsb] = ld_contig(vupT + tb);
      bun[hsb] = ld_contig(vunT + tb);
    }

    // ---- interval scores (VALU): lane owns row n15, cols hb*16+j ----
    {
      const int qg = qg_lane;
#pragma unroll 1
      for (int j = 0; j < 16; j++) {
        int kg = kb + hb * 16 + j;
        float el = 0.f, eu = 0.f;
        if (kg <= qg) {
          const float4* klr4 = (const float4*)(kl + ((size_t)bh * kT + kg) * kHS);
          const float4* kur4 = (const float4*)(ku + ((size_t)bh * kT + kg) * kHS);
          float lo = 0.f, hi = 0.f;
          auto acc1 = [&](float a, float b2, float c2, float d2) {
            float ll = a * c2, lu = a * d2, ul = b2 * c2, uu = b2 * d2;
            float mn = fminf(fminf(ll, lu), fminf(ul, uu));
            float mx = fmaxf(fmaxf(ll, lu), fmaxf(ul, uu));
            lo += fminf(fmaxf(mn, -kClamp), kClamp);
            hi += fminf(fmaxf(mx, -kClamp), kClamp);
          };
#pragma unroll
          for (int d4 = 0; d4 < 16; d4++) {
            float4 a4 = qlv[d4], b4 = quv[d4];
            float4 c4 = klr4[d4], e4 = kur4[d4];
            acc1(a4.x, b4.x, c4.x, e4.x);
            acc1(a4.y, b4.y, c4.y, e4.y);
            acc1(a4.z, b4.z, c4.z, e4.z);
            acc1(a4.w, b4.w, c4.w, e4.w);
          }
          el = __expf(lo * kScale);
          eu = __expf(hi * kScale);
        }
        ldsL[n15][hb * 16 + j] = (bf16_t)el;
        ldsU[n15][hb * 16 + j] = (bf16_t)eu;
      }
    }

    // cross-lane through LDS within one wave: split DS counter wait
    asm volatile("s_wait_dscnt 0" ::: "memory");

    // ---- read score tiles back in WMMA A-operand layout ----
    const bf16_t* srp = &ldsS[n15][0];
    const bf16_t* lrp = &ldsL[n15][0];
    const bf16_t* urp = &ldsU[n15][0];
    v16bf eaS = ld_pair(srp + d0a, srp + d0a + 16);
    v16bf eaL = ld_pair(lrp + d0a, lrp + d0a + 16);
    v16bf eaU = ld_pair(urp + d0a, urp + d0a + 16);

    // denominators: raw sums (reference uses exp without max-subtraction)
#pragma unroll
    for (int j = 0; j < 16; j++) {
      dS += (float)eaS[j];
      dL += (float)eaL[j];
      dU += (float)eaU[j];
    }

    // ---- 5 numerator GEMMs against the preloaded transposed v tiles ----
#pragma unroll
    for (int hsb = 0; hsb < 4; hsb++) {
      Ny[hsb] = wmma_bf16(eaS, bv[hsb],  Ny[hsb]);
      N1[hsb] = wmma_bf16(eaL, blp[hsb], N1[hsb]);  // e_l @ v_l+
      N2[hsb] = wmma_bf16(eaU, bln[hsb], N2[hsb]);  // e_u @ v_l-
      N3[hsb] = wmma_bf16(eaU, bup[hsb], N3[hsb]);  // e_u @ v_u+
      N4[hsb] = wmma_bf16(eaL, bun[hsb], N4[hsb]);  // e_l @ v_u-
    }
  }

  // combine half-lane partial denominators (rows live in lane&15)
  dS += __shfl_xor(dS, 16, 32);
  dL += __shfl_xor(dL, 16, 32);
  dU += __shfl_xor(dU, 16, 32);

  const int b = bh / kNH, h = bh % kNH;
#pragma unroll
  for (int r = 0; r < 8; r++) {
    int row = r + hb * 8;
    float ds_ = __shfl(dS, row, 32);
    float dl_ = __shfl(dL, row, 32);
    float du_ = __shfl(dU, row, 32);
    int mrow = b * kT + q0 + row;
#pragma unroll
    for (int hsb = 0; hsb < 4; hsb++) {
      int col = h * kHS + hsb * 16 + n15;
      float y   = Ny[hsb][r] / ds_;
      float yl  = N1[hsb][r] / du_ + N2[hsb][r] / dl_;  // s_l@v_l+ + s_u@v_l-
      float yu2 = N3[hsb][r] / dl_ + N4[hsb][r] / du_;  // s_u@v_u+ + s_l@v_u-
      size_t o = (size_t)mrow * kC + col;
      yb[o]  = (bf16_t)y;
      ycb[o] = (bf16_t)((yl + yu2) * 0.5f);  // center
      yrb[o] = (bf16_t)((yu2 - yl) * 0.5f);  // radius
    }
  }
}

// ---------------- output projection: y, y_lower, y_upper ----------------
__global__ __launch_bounds__(128) void gemm_proj_kernel(
    const bf16_t* __restrict__ ybp, const bf16_t* __restrict__ ycp,
    const bf16_t* __restrict__ yrp, const bf16_t* __restrict__ wpb,
    const bf16_t* __restrict__ wpab, const float* __restrict__ bias,
    float* __restrict__ out) {
  const int lane = threadIdx.x & 31;
  const int wv   = threadIdx.x >> 5;
  const int mb   = blockIdx.x % (kM / 64);
  const int nb   = blockIdx.x / (kM / 64);
  const int m0   = mb * 64 + wv * 16;
  const int n0   = nb * 64;
  const int n15  = lane & 15;
  const int d0a  = (lane & 16) ? 8 : 0;
  const int d0b  = (lane & 16) ? 8 : 0;
  const int mrow = m0 + n15;

  const bf16_t* yrow = ybp + (size_t)mrow * kC + d0a;
  const bf16_t* crow = ycp + (size_t)mrow * kC + d0a;
  const bf16_t* rrow = yrp + (size_t)mrow * kC + d0a;
  const bf16_t* wrow[4];
  const bf16_t* warow[4];
#pragma unroll
  for (int nblk = 0; nblk < 4; nblk++) {
    int ncol = n0 + nblk * 16 + n15;
    wrow[nblk]  = wpb  + (size_t)ncol * kC + d0b;
    warow[nblk] = wpab + (size_t)ncol * kC + d0b;
  }

  v8f aY[4], aC[4], aR[4];
#pragma unroll
  for (int i = 0; i < 4; i++) { aY[i] = vzero8(); aC[i] = vzero8(); aR[i] = vzero8(); }

  // prologue: K-block 0 -> buffer 0
  v16bf ay0 = ld_pair(yrow, yrow + 16);
  v16bf ac0 = ld_pair(crow, crow + 16);
  v16bf ar0 = ld_pair(rrow, rrow + 16);
  v16bf bw0[4], ba0[4], ay1, ac1, ar1, bw1[4], ba1[4];
#pragma unroll
  for (int nblk = 0; nblk < 4; nblk++) {
    bw0[nblk] = ld_contig(wrow[nblk]);
    ba0[nblk] = ld_contig(warow[nblk]);
  }

  for (int kb = 0; kb < kC; kb += 64) {
    const int k1 = kb + 32, k2 = kb + 64;
    // load k1 -> buffer 1
    ay1 = ld_pair(yrow + k1, yrow + k1 + 16);
    ac1 = ld_pair(crow + k1, crow + k1 + 16);
    ar1 = ld_pair(rrow + k1, rrow + k1 + 16);
#pragma unroll
    for (int nblk = 0; nblk < 4; nblk++) {
      bw1[nblk] = ld_contig(wrow[nblk] + k1);
      ba1[nblk] = ld_contig(warow[nblk] + k1);
    }
    __builtin_prefetch(wrow[0] + k2, 0, 1);
    __builtin_prefetch(warow[0] + k2, 0, 1);
    // consume buffer 0
#pragma unroll
    for (int nblk = 0; nblk < 4; nblk++) {
      aY[nblk] = wmma_bf16(ay0, bw0[nblk], aY[nblk]);
      aC[nblk] = wmma_bf16(ac0, bw0[nblk], aC[nblk]);
      aR[nblk] = wmma_bf16(ar0, ba0[nblk], aR[nblk]);
    }
    // load k2 -> buffer 0
    if (k2 < kC) {
      ay0 = ld_pair(yrow + k2, yrow + k2 + 16);
      ac0 = ld_pair(crow + k2, crow + k2 + 16);
      ar0 = ld_pair(rrow + k2, rrow + k2 + 16);
#pragma unroll
      for (int nblk = 0; nblk < 4; nblk++) {
        bw0[nblk] = ld_contig(wrow[nblk] + k2);
        ba0[nblk] = ld_contig(warow[nblk] + k2);
      }
    }
    // consume buffer 1
#pragma unroll
    for (int nblk = 0; nblk < 4; nblk++) {
      aY[nblk] = wmma_bf16(ay1, bw1[nblk], aY[nblk]);
      aC[nblk] = wmma_bf16(ac1, bw1[nblk], aC[nblk]);
      aR[nblk] = wmma_bf16(ar1, ba1[nblk], aR[nblk]);
    }
  }

  float* outY = out;
  float* outL = out + (size_t)kM * kC;
  float* outU = out + 2 * (size_t)kM * kC;
  const int rbase = (lane >> 4) * 8;
#pragma unroll
  for (int nblk = 0; nblk < 4; nblk++) {
    int col = n0 + nblk * 16 + n15;
    float bv = bias[col];
#pragma unroll
    for (int r = 0; r < 8; r++) {
      size_t o = (size_t)(m0 + rbase + r) * kC + col;
      float c = aC[nblk][r], rad = aR[nblk][r];
      outY[o] = aY[nblk][r] + bv;
      outL[o] = c - rad + bv;   // y_c@W - y_r@|W| + b
      outU[o] = c + rad + bv;   // y_c@W + y_r@|W| + b
    }
  }
}

// ---------------- host orchestration ----------------
extern "C" void kernel_launch(void* const* d_in, const int* in_sizes, int n_in,
                              void* d_out, int out_size, void* d_ws, size_t ws_size,
                              hipStream_t stream) {
  const float* x     = (const float*)d_in[0];
  const float* xerr  = (const float*)d_in[1];
  const float* Wqkv  = (const float*)d_in[2];
  const float* bqkv  = (const float*)d_in[3];
  const float* Wproj = (const float*)d_in[4];
  const float* bproj = (const float*)d_in[5];

  char* ws = (char*)d_ws;
  size_t o = 0;
  auto alloc = [&](size_t bytes) {
    size_t r = o;
    o += (bytes + 255) & ~(size_t)255;
    return r;
  };
  const size_t nMC  = (size_t)kM * kC;
  const size_t nM3C = (size_t)kM * k3C;
  const size_t nHD  = (size_t)kB * kNH * kT * kHS;  // == nMC

  bf16_t* xb   = (bf16_t*)(ws + alloc(nMC * 2));
  bf16_t* eb   = (bf16_t*)(ws + alloc(nMC * 2));
  bf16_t* WqB  = (bf16_t*)(ws + alloc((size_t)k3C * kC * 2));
  bf16_t* WqA  = (bf16_t*)(ws + alloc((size_t)k3C * kC * 2));
  bf16_t* WpB  = (bf16_t*)(ws + alloc((size_t)kC * kC * 2));
  bf16_t* WpA  = (bf16_t*)(ws + alloc((size_t)kC * kC * 2));
  float*  Cc   = (float*)(ws + alloc(nM3C * 4));
  float*  Rr   = (float*)(ws + alloc(nM3C * 4));
  bf16_t* qb   = (bf16_t*)(ws + alloc(nHD * 2));
  bf16_t* kbq  = (bf16_t*)(ws + alloc(nHD * 2));
  float*  ql   = (float*)(ws + alloc(nHD * 4));
  float*  qu   = (float*)(ws + alloc(nHD * 4));
  float*  kl   = (float*)(ws + alloc(nHD * 4));
  float*  ku   = (float*)(ws + alloc(nHD * 4));
  bf16_t* vT   = (bf16_t*)(ws + alloc(nHD * 2));
  bf16_t* vlpT = (bf16_t*)(ws + alloc(nHD * 2));
  bf16_t* vlnT = (bf16_t*)(ws + alloc(nHD * 2));
  bf16_t* vupT = (bf16_t*)(ws + alloc(nHD * 2));
  bf16_t* vunT = (bf16_t*)(ws + alloc(nHD * 2));
  bf16_t* yb   = (bf16_t*)(ws + alloc(nMC * 2));
  bf16_t* ycb  = (bf16_t*)(ws + alloc(nMC * 2));
  bf16_t* yrb  = (bf16_t*)(ws + alloc(nMC * 2));

  // 1. precision conversion (+abs weights for radius GEMMs)
  cvt_kernel<<<(int)((nMC + 255) / 256), 256, 0, stream>>>(x, xb, (int)nMC);
  cvt_kernel<<<(int)((nMC + 255) / 256), 256, 0, stream>>>(xerr, eb, (int)nMC);
  cvt_abs_kernel<<<(int)(((size_t)k3C * kC + 255) / 256), 256, 0, stream>>>(
      Wqkv, WqB, WqA, k3C * kC);
  cvt_abs_kernel<<<(int)(((size_t)kC * kC + 255) / 256), 256, 0, stream>>>(
      Wproj, WpB, WpA, kC * kC);

  // 2. fused QKV center/radius GEMM (WMMA bf16, ping-pong K loop)
  gemm_qkv_kernel<<<(kM / 64) * (k3C / 64), 128, 0, stream>>>(
      xb, eb, WqB, WqA, bqkv, Cc, Rr);

  // 3. repack into head-major q/k bounds + transposed v splits
  repack_kernel<<<(int)((nM3C + 255) / 256), 256, 0, stream>>>(
      Cc, Rr, qb, ql, qu, kbq, kl, ku, vT, vlpT, vlnT, vupT, vunT);

  // 4. fused causal std+interval attention (single pass, 1 wave / workgroup)
  attn_kernel<<<(kB * kNH) * (kT / 16), 32, 0, stream>>>(
      qb, kbq, ql, qu, kl, ku, vT, vlpT, vlnT, vupT, vunT, yb, ycb, yrb);

  // 5. projection -> (y, y_lower, y_upper) concatenated into d_out
  gemm_proj_kernel<<<(kM / 64) * (kC / 64), 128, 0, stream>>>(
      yb, ycb, yrb, WpB, WpA, bproj, (float*)d_out);
}
